// ActorTanh_67327907332481
// MI455X (gfx1250) — matrix operand order repaired and checked by reading, not compiled, not run on hardware
//
#include <hip/hip_runtime.h>
#include <hip/hip_bf16.h>
#include <math.h>

#define BS    1024
#define NB    30
#define KNN   10
#define HID   128
#define EMB   64
#define NNODE (BS * NB)            // 30720
#define FEATC (HID + EMB)          // 192
#define ECOLS (2 * FEATC)          // 384

typedef __attribute__((ext_vector_type(16))) _Float16 v16h;
typedef __attribute__((ext_vector_type(8)))  _Float16 v8h;
typedef __attribute__((ext_vector_type(8)))  float    v8f;

// ---------------------------------------------------------------------------
// Branch-free native transcendentals (v_exp_f32 / v_rcp_f32), no ocml calls.
// tanh(x) = 1 - 2/(e^{2x}+1); exp2 saturates (inf / 0) so the limits +-1 are
// produced automatically without branches.
// ---------------------------------------------------------------------------
__device__ __forceinline__ float tanh_fast(float x) {
  float e = __builtin_amdgcn_exp2f(x * 2.885390081777926814f);   // 2*log2(e)
  return 1.0f - 2.0f * __builtin_amdgcn_rcpf(e + 1.0f);
}
__device__ __forceinline__ float exp_fast(float x) {
  return __builtin_amdgcn_exp2f(x * 1.442695040888963407f);      // log2(e)
}

// ---------------------------------------------------------------------------
// WMMA fragment helpers (wave32, v_wmma_f32_16x16x32_f16)
//
// A: 16x32 (MxK) row-major source.  Lane half h holds 16 contiguous
// K: [kbase+h*16, +16), matching frag_b_sw.
// Loaded as two aligned 16-byte vectors -> 2x ds/global_load_b128.
// ---------------------------------------------------------------------------
__device__ __forceinline__ v16h frag_a(const _Float16* base, int ldm,
                                       int m, int half, int kbase) {
  const _Float16* row = base + (size_t)m * ldm + kbase + half * 16;
  v8h lo = *(const v8h*)(row);        // K = kbase + half*16 + [0,8)
  v8h hi = *(const v8h*)(row + 8);    // K = kbase + half*16 + [8,16)
  v16h a;
#pragma unroll
  for (int j = 0; j < 8; ++j) { a[j] = lo[j]; a[8 + j] = hi[j]; }
  return a;
}

// B: 32x16 (KxN) from a fragment-swizzled weight buffer:
//   Wsw[((k/16)*N + n)*16 + (k%16)] = W[k][n]
// Lane (half,n) fragment = 16 contiguous halfs (K = kbase + half*16 + [0,16)),
// element i of the vector <-> K = k0 + i, the same contiguous k order as frag_a.
// One aligned 32-byte load -> 2x global_load_b128.
// ---------------------------------------------------------------------------
__device__ __forceinline__ v16h frag_b_sw(const _Float16* Wsw, int N,
                                          int n, int half, int kbase) {
  const _Float16* p = Wsw + ((size_t)((kbase >> 4) + half) * N + n) * 16;
  return *(const v16h*)p;
}

#define WMMA_F16(a, b, c) \
  __builtin_amdgcn_wmma_f32_16x16x32_f16(false, (a), false, (b), (short)0, (c), false, false)

// ---------------------------------------------------------------------------
// K1: per-batch KNN.  1024 blocks x 32 threads (one wave). Lane i < 30 owns
// node i; selects K=10 nearest (lowest index on ties, matching top_k(-d2)).
// ---------------------------------------------------------------------------
__global__ __launch_bounds__(32) void k_nbr(const float* __restrict__ state,
                                            int* __restrict__ nbr) {
  int b = blockIdx.x;
  __shared__ float px[NB], py[NB];
  int lane = threadIdx.x;
  if (lane < NB) {
    px[lane] = state[b * (2 * NB) + 2 * lane];
    py[lane] = state[b * (2 * NB) + 2 * lane + 1];
  }
  __syncthreads();
  if (lane >= NB) return;
  float d2[NB];
  float xi = px[lane], yi = py[lane];
#pragma unroll
  for (int j = 0; j < NB; ++j) {
    float dx = xi - px[j], dy = yi - py[j];
    d2[j] = dx * dx + dy * dy;
  }
  d2[lane] += 1.0e10f;
  unsigned mask = 0;
  int* out = nbr + ((size_t)b * NB + lane) * KNN;
  for (int kk = 0; kk < KNN; ++kk) {
    float best = 3.4e38f;
    int bi = 0;
#pragma unroll
    for (int j = 0; j < NB; ++j) {
      bool take = !((mask >> j) & 1u) && (d2[j] < best);
      best = take ? d2[j] : best;
      bi   = take ? j : bi;
    }
    mask |= 1u << bi;
    out[kk] = bi;
  }
}

// ---------------------------------------------------------------------------
// K2: weight f32 -> f16 conversion into WMMA-fragment-swizzled layout
//     + batch-invariant class features cf[i][e] = tanh(tanh(emb[i/10])@W_emb+b)
// ---------------------------------------------------------------------------
__device__ __forceinline__ void cvt_swizzle(const float* __restrict__ W,
                                            _Float16* __restrict__ Wsw,
                                            int Kdim, int t, int stride) {
  for (int i = t; i < Kdim * HID; i += stride) {
    int k = i >> 7, n = i & 127;
    Wsw[(((size_t)(k >> 4) * HID + n) << 4) + (k & 15)] = (_Float16)W[i];
  }
}

__global__ void k_prep(const float* __restrict__ W_in2, const float* __restrict__ W_m1,
                       const float* __restrict__ W_m2, const float* __restrict__ W_a1,
                       const float* __restrict__ emb_tab, const float* __restrict__ W_emb,
                       const float* __restrict__ b_emb,
                       _Float16* __restrict__ w_in2h, _Float16* __restrict__ w_m1h,
                       _Float16* __restrict__ w_m2h, _Float16* __restrict__ w_a1h,
                       _Float16* __restrict__ cf) {
  int t = blockIdx.x * blockDim.x + threadIdx.x;
  int stride = gridDim.x * blockDim.x;
  cvt_swizzle(W_in2, w_in2h, HID,   t, stride);
  cvt_swizzle(W_m1,  w_m1h,  ECOLS, t, stride);
  cvt_swizzle(W_m2,  w_m2h,  HID,   t, stride);
  cvt_swizzle(W_a1,  w_a1h,  HID,   t, stride);
  for (int i = t; i < NB * EMB; i += stride) {
    int node = i >> 6, e = i & 63;
    int c = node / (NB / 3);
    float acc = b_emb[e];
#pragma unroll 8
    for (int k = 0; k < EMB; ++k)
      acc += tanh_fast(emb_tab[c * EMB + k]) * W_emb[k * EMB + e];
    cf[i] = (_Float16)tanh_fast(acc);
  }
}

// ---------------------------------------------------------------------------
// K3: feat MLP.  480 blocks x 256 threads, 64 node-rows per block.
// layer1 (K=4) in VALU -> LDS f16; layer2 (128x128) via WMMA; tanh epilogue.
// ---------------------------------------------------------------------------
__global__ __launch_bounds__(256) void k_feat(const float* __restrict__ state,
                                              const float* __restrict__ tar,
                                              const float* __restrict__ W_in1,
                                              const float* __restrict__ b_in1,
                                              const _Float16* __restrict__ W_in2h,
                                              const float* __restrict__ b_in2,
                                              const _Float16* __restrict__ cf,
                                              _Float16* __restrict__ feat) {
  __shared__ alignas(16) _Float16 h1[64][HID];   // 16 KB
  int r0 = blockIdx.x * 64;
  int tid = threadIdx.x;

  // Phase A: h1 = tanh(inp4 @ W_in1 + b_in1)
  {
    int mrow = tid >> 2;           // 0..63
    int csub = tid & 3;            // 32 cols each
    int r = r0 + mrow;
    float p0 = state[(size_t)r * 2];
    float p1 = state[(size_t)r * 2 + 1];
    float t0 = tanh_fast(tar[(size_t)r * 2]);
    float t1 = tanh_fast(tar[(size_t)r * 2 + 1]);
#pragma unroll 4
    for (int c = csub * 32; c < csub * 32 + 32; ++c) {
      float acc = b_in1[c] + p0 * W_in1[c] + p1 * W_in1[HID + c] +
                  t0 * W_in1[2 * HID + c] + t1 * W_in1[3 * HID + c];
      h1[mrow][c] = (_Float16)tanh_fast(acc);
    }
  }
  __syncthreads();

  // Phase B: feat[:, :128] = tanh(h1 @ W_in2 + b_in2) via WMMA
  int wave = tid >> 5, lane = tid & 31;
  int half = lane >> 4, nl = lane & 15, ml = lane & 15;
  int mt = wave >> 1;
#pragma unroll
  for (int ntq = 0; ntq < 4; ++ntq) {
    int nt = (wave & 1) * 4 + ntq;
    v8f acc = {};
#pragma unroll
    for (int kc = 0; kc < 4; ++kc) {
      v16h a = frag_a(&h1[0][0], HID, mt * 16 + ml, half, kc * 32);
      v16h b = frag_b_sw(W_in2h, HID, nt * 16 + nl, half, kc * 32);
      acc = WMMA_F16(a, b, acc);
    }
    int col = nt * 16 + nl;
    float bias = b_in2[col];
#pragma unroll
    for (int rr = 0; rr < 8; ++rr) {
      int rloc = mt * 16 + rr + half * 8;
      feat[(size_t)(r0 + rloc) * FEATC + col] = (_Float16)tanh_fast(acc[rr] + bias);
    }
  }

  // Phase C: feat[:, 128:192] = tanh(class_feat)  (precomputed)
  for (int i = tid; i < 64 * EMB; i += 256) {
    int rr = i >> 6, e = i & 63;
    int node = (r0 + rr) % NB;
    feat[(size_t)(r0 + rr) * FEATC + HID + e] = cf[node * EMB + e];
  }
}

// ---------------------------------------------------------------------------
// K4: message MLP + max-reduce.  7680 blocks x 256 threads; 4 nodes (40 e-rows,
// padded to 48) per block.  48 KB static LDS.
// ---------------------------------------------------------------------------
__global__ __launch_bounds__(256) void k_msg(const _Float16* __restrict__ feat,
                                             const int* __restrict__ nbr,
                                             const _Float16* __restrict__ Wm1h,
                                             const float* __restrict__ b_m1,
                                             const _Float16* __restrict__ Wm2h,
                                             const float* __restrict__ b_m2,
                                             _Float16* __restrict__ xout) {
  __shared__ alignas(16) _Float16 eLDS[48][ECOLS];  // 36 KB (reused as f32 msg)
  __shared__ alignas(16) _Float16 hLDS[48][HID];    // 12 KB
  int tid = threadIdx.x;
  int node0 = blockIdx.x * 4;   // global node row base

  // Phase A: build e = [x_i , x_j - x_i] in LDS (rows 40..47 zero pad).
  // 8-half (16 B) vector chunks; FEATC=192 is a multiple of 8 so no chunk
  // straddles the concat boundary.
  for (int idx = tid; idx < 48 * 48; idx += 256) {
    int rr = idx / 48;
    int c8 = (idx % 48) * 8;
    v8h outv = {};
    if (rr < 40) {
      int gi = node0 + rr / KNN;          // global node index (b*NB + i)
      int j = rr % KNN;
      int bb = gi / NB;
      int nloc = nbr[(size_t)gi * KNN + j];
      const _Float16* fi = feat + (size_t)gi * FEATC;
      const _Float16* fj = feat + ((size_t)bb * NB + nloc) * FEATC;
      if (c8 < FEATC) {
        outv = *(const v8h*)(fi + c8);
      } else {
        v8h a = *(const v8h*)(fj + (c8 - FEATC));
        v8h b = *(const v8h*)(fi + (c8 - FEATC));
        outv = a - b;
      }
    }
    *(v8h*)&eLDS[rr][c8] = outv;
  }
  __syncthreads();

  int wave = tid >> 5, lane = tid & 31;
  int half = lane >> 4, nl = lane & 15, ml = lane & 15;
  int nt = wave;                 // 8 waves cover 8 n-tiles; each does mt=0..2
  int col = nt * 16 + nl;

  // Phase B: GEMM1  h = tanh(e @ W_m1 + b_m1)   K = 384 (12 chunks)
  {
    v8f acc[3] = {{}, {}, {}};
#pragma unroll
    for (int kc = 0; kc < 12; ++kc) {
      v16h b = frag_b_sw(Wm1h, HID, col, half, kc * 32);
#pragma unroll
      for (int mt = 0; mt < 3; ++mt) {
        v16h a = frag_a(&eLDS[0][0], ECOLS, mt * 16 + ml, half, kc * 32);
        acc[mt] = WMMA_F16(a, b, acc[mt]);
      }
    }
    float bias = b_m1[col];
#pragma unroll
    for (int mt = 0; mt < 3; ++mt)
#pragma unroll
      for (int rr = 0; rr < 8; ++rr)
        hLDS[mt * 16 + rr + half * 8][col] = (_Float16)tanh_fast(acc[mt][rr] + bias);
  }
  __syncthreads();   // all e reads + h writes complete

  // Phase C: GEMM2  msg = h @ W_m2 + b_m2   (msg f32 overlays eLDS region)
  float* msg = (float*)&eLDS[0][0];   // 48*128 f32 = 24 KB <= 36 KB
  {
    v8f acc[3] = {{}, {}, {}};
#pragma unroll
    for (int kc = 0; kc < 4; ++kc) {
      v16h b = frag_b_sw(Wm2h, HID, col, half, kc * 32);
#pragma unroll
      for (int mt = 0; mt < 3; ++mt) {
        v16h a = frag_a(&hLDS[0][0], HID, mt * 16 + ml, half, kc * 32);
        acc[mt] = WMMA_F16(a, b, acc[mt]);
      }
    }
    float bias = b_m2[col];
#pragma unroll
    for (int mt = 0; mt < 3; ++mt)
#pragma unroll
      for (int rr = 0; rr < 8; ++rr)
        msg[(size_t)(mt * 16 + rr + half * 8) * HID + col] = acc[mt][rr] + bias;
  }
  __syncthreads();

  // Phase D: x = tanh(max over K=10 messages)
  for (int idx = tid; idx < 4 * HID; idx += 256) {
    int g = idx >> 7, c = idx & 127;
    float mx = -3.4e38f;
#pragma unroll
    for (int r = 0; r < KNN; ++r)
      mx = fmaxf(mx, msg[(size_t)(g * KNN + r) * HID + c]);
    xout[(size_t)(node0 + g) * HID + c] = (_Float16)tanh_fast(mx);
  }
}

// ---------------------------------------------------------------------------
// K5: actor head + postprocess.  480 blocks x 256 threads, 64 rows per block.
// ---------------------------------------------------------------------------
__global__ __launch_bounds__(256) void k_out(const _Float16* __restrict__ x,
                                             const _Float16* __restrict__ Wa1h,
                                             const float* __restrict__ b_a1,
                                             const float* __restrict__ W_a2,
                                             const float* __restrict__ b_a2,
                                             const float* __restrict__ tar,
                                             float* __restrict__ out) {
  __shared__ alignas(16) _Float16 y[64][HID];   // 16 KB
  int r0 = blockIdx.x * 64;
  int tid = threadIdx.x;
  int wave = tid >> 5, lane = tid & 31;
  int half = lane >> 4, nl = lane & 15, ml = lane & 15;
  int mt = wave >> 1;

  // y = tanh(x @ W_a1 + b_a1), A fragments straight from global f16 x
  const _Float16* xbase = x + (size_t)r0 * HID;
#pragma unroll
  for (int ntq = 0; ntq < 4; ++ntq) {
    int nt = (wave & 1) * 4 + ntq;
    v8f acc = {};
#pragma unroll
    for (int kc = 0; kc < 4; ++kc) {
      v16h a = frag_a(xbase, HID, mt * 16 + ml, half, kc * 32);
      v16h b = frag_b_sw(Wa1h, HID, nt * 16 + nl, half, kc * 32);
      acc = WMMA_F16(a, b, acc);
    }
    int col = nt * 16 + nl;
    float bias = b_a1[col];
#pragma unroll
    for (int rr = 0; rr < 8; ++rr)
      y[mt * 16 + rr + half * 8][col] = (_Float16)tanh_fast(acc[rr] + bias);
  }
  __syncthreads();

  // 128 -> 4 head + mu/std postprocess; one (row, out-col) per thread
  int rr = tid >> 2, c = tid & 3;
  int r = r0 + rr;
  float acc = b_a2[c];
#pragma unroll 8
  for (int k = 0; k < HID; ++k)
    acc += (float)y[rr][k] * W_a2[k * 4 + c];
  int b = r / NB, i = r % NB;
  if (c < 2) {
    float mu = 0.3f * tanh_fast(acc) + 0.3f * tanh_fast(tar[(size_t)r * 2 + c]);
    out[(size_t)b * (2 * NB) + 2 * i + c] = mu;
  } else {
    float ls = tanh_fast(acc);
    ls = -5.0f + 0.5f * 7.0f * (ls + 1.0f);
    out[(size_t)BS * (2 * NB) + (size_t)b * (2 * NB) + 2 * i + (c - 2)] = exp_fast(ls);
  }
}

// ---------------------------------------------------------------------------
extern "C" void kernel_launch(void* const* d_in, const int* in_sizes, int n_in,
                              void* d_out, int out_size, void* d_ws, size_t ws_size,
                              hipStream_t stream) {
  (void)in_sizes; (void)n_in; (void)out_size; (void)ws_size;
  const float* state = (const float*)d_in[0];
  const float* tar   = (const float*)d_in[1];
  const float* W_in1 = (const float*)d_in[2];
  const float* b_in1 = (const float*)d_in[3];
  const float* W_in2 = (const float*)d_in[4];
  const float* b_in2 = (const float*)d_in[5];
  const float* emb   = (const float*)d_in[6];
  const float* W_emb = (const float*)d_in[7];
  const float* b_emb = (const float*)d_in[8];
  const float* W_m1  = (const float*)d_in[9];
  const float* b_m1  = (const float*)d_in[10];
  const float* W_m2  = (const float*)d_in[11];
  const float* b_m2  = (const float*)d_in[12];
  const float* W_a1  = (const float*)d_in[13];
  const float* b_a1  = (const float*)d_in[14];
  const float* W_a2  = (const float*)d_in[15];
  const float* b_a2  = (const float*)d_in[16];

  char* ws = (char*)d_ws;
  size_t off = 0;
  auto carve = [&](size_t bytes) -> void* {
    void* p = ws + off;
    off += (bytes + 255) & ~(size_t)255;
    return p;
  };
  int*      nbr    = (int*)     carve((size_t)NNODE * KNN * sizeof(int));
  _Float16* feat   = (_Float16*)carve((size_t)NNODE * FEATC * sizeof(_Float16));
  _Float16* xbuf   = (_Float16*)carve((size_t)NNODE * HID * sizeof(_Float16));
  _Float16* cf     = (_Float16*)carve((size_t)NB * EMB * sizeof(_Float16));
  _Float16* w_in2h = (_Float16*)carve((size_t)HID * HID * sizeof(_Float16));
  _Float16* w_m1h  = (_Float16*)carve((size_t)ECOLS * HID * sizeof(_Float16));
  _Float16* w_m2h  = (_Float16*)carve((size_t)HID * HID * sizeof(_Float16));
  _Float16* w_a1h  = (_Float16*)carve((size_t)HID * HID * sizeof(_Float16));

  k_nbr <<<BS, 32, 0, stream>>>(state, nbr);
  k_prep<<<128, 256, 0, stream>>>(W_in2, W_m1, W_m2, W_a1, emb, W_emb, b_emb,
                                  w_in2h, w_m1h, w_m2h, w_a1h, cf);
  k_feat<<<NNODE / 64, 256, 0, stream>>>(state, tar, W_in1, b_in1,
                                         w_in2h, b_in2, cf, feat);
  k_msg <<<NNODE / 4, 256, 0, stream>>>(feat, nbr, w_m1h, b_m1, w_m2h, b_m2, xbuf);
  k_out <<<NNODE / 64, 256, 0, stream>>>(xbuf, w_a1h, b_a1, W_a2, b_a2, tar,
                                         (float*)d_out);
}
